// SpatialTransformerDecoder_15985868276192
// MI455X (gfx1250) — matrix operand order repaired and checked by reading, not compiled, not run on hardware
//
#include <hip/hip_runtime.h>
#include <hip/hip_bf16.h>

// ---------------------------------------------------------------------------
// SpatialTransformerDecoder for MI455X (gfx1250, wave32, WMMA)
//   B=512, C=4, N=24, CN=96, D=512, H=8 (dh=64), L=4, DFF=2048, D2=256, D4=128
// Dead code (depth encoder / memory projection) is skipped entirely.
// Big GEMMs (~1.07 TFLOP) run through v_wmma_f32_16x16x32_bf16 with LDS tiles
// stored in fragment layout (ds_load_b128 feeds); the 4-key attention,
// LayerNorms and tiny head dots run as wave32 VALU kernels.
// ---------------------------------------------------------------------------

#define BB 512
#define CC 4
#define NN 24
#define CN 96            // C*N
#define DD 512
#define HH 8
#define DH 64
#define LL 4
#define DFF 2048
#define D2 256
#define D4 128
#define ROWS (BB * CN)   // 49152
#define MROWS (BB * CC)  // 2048 (memory rows for K/V)
#define FFN_CHUNK 4096   // M-chunk for FFN hidden scratch

typedef __bf16 v16bf __attribute__((ext_vector_type(16)));
typedef float  v8f   __attribute__((ext_vector_type(8)));

__device__ __forceinline__ float gelu_f(float x) {
    // tanh approximation (jax.nn.gelu default)
    float x3 = x * x * x;
    return 0.5f * x * (1.0f + tanhf(0.7978845608028654f * (x + 0.044715f * x3)));
}

__device__ __forceinline__ float softplus_f(float x) {
    return (x > 20.0f) ? x : log1pf(__expf(x));
}

__device__ __forceinline__ float wave_sum(float v) {
    #pragma unroll
    for (int off = 16; off > 0; off >>= 1) v += __shfl_xor(v, off, 32);
    return v;
}

// ---------------------------------------------------------------------------
// q0[b, c*N+n, :] = view_emb[c,:] + kp_emb[n,:]
// ---------------------------------------------------------------------------
__global__ __launch_bounds__(256) void k_build_q(const float* __restrict__ view_emb,
                                                 const float* __restrict__ kp_emb,
                                                 float* __restrict__ q) {
    long idx = (long)blockIdx.x * 256 + threadIdx.x;  // over ROWS*D
    int d  = (int)(idx % DD);
    long t = idx / DD;
    int cn = (int)(t % CN);
    int c  = cn / NN;
    int n  = cn % NN;
    q[idx] = view_emb[c * DD + d] + kp_emb[n * DD + d];
}

// ---------------------------------------------------------------------------
// Generic GEMM: Cout[M,N] = act(A[M,K] @ W[K,N] + bias[N])
// Block tile: 128x64 (256 threads = 8 wave32 waves). Each wave owns a 2x2 set
// of 16x16 subtiles (4 fp32 accumulators), K stepped by 32.
//
// LDS tiles are stored in WMMA *fragment layout*: each (subtile, lane) slot is
// 16 bf16 = 32 contiguous bytes, so a fragment load is two ds_load_b128.
//   A frag (16x32):  lanes 0-15 -> K 0..7 & 16..23 ; lanes 16-31 -> K 8..15 & 24..31
//   B frag (32x16):  lanes 0-15 -> K 0..7 & 16..23 ; lanes 16-31 -> K 8..15 & 24..31
//
// Requires: M%128==0, N%64==0, K%32==0 (true for every call here).
// act: 0 = none, 1 = GELU
// ---------------------------------------------------------------------------
__global__ __launch_bounds__(256) void k_gemm(const float* __restrict__ A,
                                              const float* __restrict__ W,
                                              const float* __restrict__ bias,
                                              float* __restrict__ Cout,
                                              int M, int N, int K, int act) {
    __shared__ v16bf Asf[8 * 32];   // 8 M-subtiles x 32 lanes  (128x32 A tile, 8KB)
    __shared__ v16bf Bsf[4 * 32];   // 4 N-subtiles x 32 lanes  (32x64  B tile, 4KB)
    uint4* As4 = (uint4*)Asf;
    uint4* Bs4 = (uint4*)Bsf;

    const int tid  = threadIdx.x;
    const int wave = tid >> 5;
    const int lane = tid & 31;
    const int mi2  = (wave & 3) * 2;   // M subtile pair base (0,2,4,6)
    const int nt2  = (wave >> 2) * 2;  // N subtile pair base (0,2)

    const int tileM = blockIdx.y * 128;
    const int tileN = blockIdx.x * 64;

    v8f acc00 = {}, acc01 = {}, acc10 = {}, acc11 = {};

    // A staging coords: thread -> row r (0..127), K-half (0 or 16)
    const int ar   = tid >> 1;
    const int ahal = tid & 1;           // 0: K 0..15, 1: K 16..31
    const int ami  = ar >> 4;           // M subtile 0..7
    const int am   = ar & 15;
    // B staging coords: thread -> col n (0..63), K-segment (0..3)
    const int bn   = tid & 63;
    const int bks  = tid >> 6;          // 8-wide K segment
    const int bnt  = bn >> 4;
    const int bl   = (bn & 15) + ((bks & 1) ? 16 : 0);
    const int bidx = ((bnt * 32 + bl) << 1) + (bks >> 1);

    for (int k0 = 0; k0 < K; k0 += 32) {
        __syncthreads();
        {   // stage A: 16 contiguous floats per thread (4x global_load_b128)
            const float4* src =
                (const float4*)(A + (size_t)(tileM + ar) * K + k0 + ahal * 16);
            float4 f0 = src[0], f1 = src[1], f2 = src[2], f3 = src[3];
            union { __bf16 h[8]; uint4 v; } u0, u1;
            u0.h[0] = (__bf16)f0.x; u0.h[1] = (__bf16)f0.y;
            u0.h[2] = (__bf16)f0.z; u0.h[3] = (__bf16)f0.w;
            u0.h[4] = (__bf16)f1.x; u0.h[5] = (__bf16)f1.y;
            u0.h[6] = (__bf16)f1.z; u0.h[7] = (__bf16)f1.w;
            u1.h[0] = (__bf16)f2.x; u1.h[1] = (__bf16)f2.y;
            u1.h[2] = (__bf16)f2.z; u1.h[3] = (__bf16)f2.w;
            u1.h[4] = (__bf16)f3.x; u1.h[5] = (__bf16)f3.y;
            u1.h[6] = (__bf16)f3.z; u1.h[7] = (__bf16)f3.w;
            // chunk j=0: K base +0 -> lane am      ; chunk j=1: K base +8 -> lane am+16
            As4[((ami * 32 + am) << 1) + ahal]      = u0.v;
            As4[((ami * 32 + am + 16) << 1) + ahal] = u1.v;
        }
        {   // stage B: 8-long K-column segment per thread (coalesced across lanes)
            const float* src = W + (size_t)(k0 + bks * 8) * N + tileN + bn;
            union { __bf16 h[8]; uint4 v; } u;
            #pragma unroll
            for (int i = 0; i < 8; ++i) u.h[i] = (__bf16)src[(size_t)i * N];
            Bs4[bidx] = u.v;
        }
        __syncthreads();

        // fragment loads: 2x ds_load_b128 each
        v16bf af0 = Asf[(mi2 + 0) * 32 + lane];
        v16bf af1 = Asf[(mi2 + 1) * 32 + lane];
        v16bf bf0 = Bsf[(nt2 + 0) * 32 + lane];
        v16bf bf1 = Bsf[(nt2 + 1) * 32 + lane];

        acc00 = __builtin_amdgcn_wmma_f32_16x16x32_bf16(false, af0, false, bf0,
                                                        (short)0, acc00, false, false);
        acc01 = __builtin_amdgcn_wmma_f32_16x16x32_bf16(false, af0, false, bf1,
                                                        (short)0, acc01, false, false);
        acc10 = __builtin_amdgcn_wmma_f32_16x16x32_bf16(false, af1, false, bf0,
                                                        (short)0, acc10, false, false);
        acc11 = __builtin_amdgcn_wmma_f32_16x16x32_bf16(false, af1, false, bf1,
                                                        (short)0, acc11, false, false);
    }

    // Epilogue: C layout — VGPR r: lanes0-15 M=r, lanes16-31 M=r+8; N=lane&15
    const int ncol = lane & 15;
    const int mofs = (lane >> 4) * 8;
    const int col0 = tileN + (nt2 + 0) * 16 + ncol;
    const int col1 = tileN + (nt2 + 1) * 16 + ncol;
    const float b0 = bias[col0];
    const float b1 = bias[col1];
    #pragma unroll
    for (int i = 0; i < 2; ++i) {
        const v8f& a0 = (i == 0) ? acc00 : acc10;
        const v8f& a1 = (i == 0) ? acc01 : acc11;
        int rowb = tileM + (mi2 + i) * 16 + mofs;
        #pragma unroll
        for (int r = 0; r < 8; ++r) {
            float v0 = a0[r] + b0;
            float v1 = a1[r] + b1;
            if (act == 1) { v0 = gelu_f(v0); v1 = gelu_f(v1); }
            Cout[(size_t)(rowb + r) * N + col0] = v0;
            Cout[(size_t)(rowb + r) * N + col1] = v1;
        }
    }
}

// ---------------------------------------------------------------------------
// Cross-attention core: K-dim is only C=4 keys -> pure VALU, 1 wave per (b,q,h)
// qh: (B,CN,D) as (B,CN,H,dh); kh/vh: (B,C,D); out context: (B,CN,D)
// ---------------------------------------------------------------------------
__global__ __launch_bounds__(256) void k_attn(const float* __restrict__ qh,
                                              const float* __restrict__ kh,
                                              const float* __restrict__ vh,
                                              float* __restrict__ outc) {
    int gw   = (blockIdx.x * 256 + threadIdx.x) >> 5;   // wave id: B*CN*H total
    int lane = threadIdx.x & 31;
    int h  = gw % HH;
    int t  = gw / HH;
    int qi = t % CN;
    int b  = t / CN;

    const float* qp = qh + (size_t)(b * CN + qi) * DD + h * DH;
    float q0 = qp[lane], q1 = qp[lane + 32];

    float s[CC];
    #pragma unroll
    for (int c = 0; c < CC; ++c) {
        const float* kp = kh + (size_t)(b * CC + c) * DD + h * DH;
        float p = q0 * kp[lane] + q1 * kp[lane + 32];
        s[c] = wave_sum(p) * 0.125f;   // 1/sqrt(64)
    }
    float m = fmaxf(fmaxf(s[0], s[1]), fmaxf(s[2], s[3]));
    float e[CC];
    float sum = 0.0f;
    #pragma unroll
    for (int c = 0; c < CC; ++c) { e[c] = __expf(s[c] - m); sum += e[c]; }
    float inv = 1.0f / sum;

    float o0 = 0.0f, o1 = 0.0f;
    #pragma unroll
    for (int c = 0; c < CC; ++c) {
        const float* vp = vh + (size_t)(b * CC + c) * DD + h * DH;
        float a = e[c] * inv;
        o0 += a * vp[lane];
        o1 += a * vp[lane + 32];
    }
    float* op = outc + (size_t)(b * CN + qi) * DD + h * DH;
    op[lane] = o0;
    op[lane + 32] = o1;
}

// ---------------------------------------------------------------------------
// q = LayerNorm(q + x) * g + b     (row length D=512, 1 wave per row)
// ---------------------------------------------------------------------------
__global__ __launch_bounds__(256) void k_res_ln(float* __restrict__ q,
                                                const float* __restrict__ x,
                                                const float* __restrict__ g,
                                                const float* __restrict__ bta,
                                                int rows) {
    int gw   = (blockIdx.x * 256 + threadIdx.x) >> 5;
    int lane = threadIdx.x & 31;
    if (gw >= rows) return;
    float* qr = q + (size_t)gw * DD;
    const float* xr = x + (size_t)gw * DD;

    float v[16];
    float s = 0.0f;
    #pragma unroll
    for (int i = 0; i < 16; ++i) {
        int d = lane + i * 32;
        v[i] = qr[d] + xr[d];
        s += v[i];
    }
    float mean = wave_sum(s) * (1.0f / (float)DD);
    float var = 0.0f;
    #pragma unroll
    for (int i = 0; i < 16; ++i) { float t = v[i] - mean; var += t * t; }
    var = wave_sum(var) * (1.0f / (float)DD);
    float inv = rsqrtf(var + 1e-5f);
    #pragma unroll
    for (int i = 0; i < 16; ++i) {
        int d = lane + i * 32;
        qr[d] = (v[i] - mean) * inv * g[d] + bta[d];
    }
}

// ---------------------------------------------------------------------------
// Final heads: 1 wave per row. pred_w3 (128x2), dp_w3 (128x1), conf_w (512x1),
// then occlusion selection into the three outputs.
// ---------------------------------------------------------------------------
__global__ __launch_bounds__(256) void k_final(const float* __restrict__ feat,
                                               const float* __restrict__ P2,
                                               const float* __restrict__ Q2,
                                               const float* __restrict__ pw3,
                                               const float* __restrict__ pb3,
                                               const float* __restrict__ dw3,
                                               const float* __restrict__ db3,
                                               const float* __restrict__ cw,
                                               const float* __restrict__ cb,
                                               const float* __restrict__ okp,
                                               const int* __restrict__ occ,
                                               float* __restrict__ out_coords,
                                               float* __restrict__ out_conf,
                                               float* __restrict__ out_depth) {
    int gw   = (blockIdx.x * 256 + threadIdx.x) >> 5;  // row id, ROWS total
    int lane = threadIdx.x & 31;

    const float* fr = feat + (size_t)gw * DD;
    float cs = 0.0f;
    #pragma unroll
    for (int i = 0; i < 16; ++i) { int d = lane + i * 32; cs += fr[d] * cw[d]; }
    cs = wave_sum(cs);

    const float* pr = P2 + (size_t)gw * D4;
    float px = 0.0f, py = 0.0f;
    #pragma unroll
    for (int i = 0; i < 4; ++i) {
        int k = lane + i * 32;
        float pv = pr[k];
        px += pv * pw3[k * 2];
        py += pv * pw3[k * 2 + 1];
    }
    px = wave_sum(px);
    py = wave_sum(py);

    const float* qr = Q2 + (size_t)gw * D4;
    float dd = 0.0f;
    #pragma unroll
    for (int i = 0; i < 4; ++i) { int k = lane + i * 32; dd += qr[k] * dw3[k]; }
    dd = wave_sum(dd);

    if (lane == 0) {
        bool occluded = (occ[gw] == 0);   // mask True = visible
        float cx, cy;
        if (occluded) { cx = px + pb3[0]; cy = py + pb3[1]; }
        else          { cx = okp[(size_t)gw * 2]; cy = okp[(size_t)gw * 2 + 1]; }
        out_coords[(size_t)gw * 2]     = cx;
        out_coords[(size_t)gw * 2 + 1] = cy;
        out_conf[gw]  = occluded ? 1.0f / (1.0f + __expf(-(cs + cb[0]))) : 0.0f;
        out_depth[gw] = occluded ? softplus_f(dd + db3[0]) : 0.0f;
    }
}

// ---------------------------------------------------------------------------
// Host orchestration
// ---------------------------------------------------------------------------
extern "C" void kernel_launch(void* const* d_in, const int* in_sizes, int n_in,
                              void* d_out, int out_size, void* d_ws, size_t ws_size,
                              hipStream_t stream) {
    (void)in_sizes; (void)n_in; (void)out_size; (void)ws_size;

    const float* enc   = (const float*)d_in[0];   // (B,C,D)
    const float* okp   = (const float*)d_in[1];   // (B,C,N,2)
    // d_in[2] = depth_map  : dead w.r.t. outputs
    const int*   occ   = (const int*)  d_in[3];   // (B,C,N) bool->int
    const float* kp_e  = (const float*)d_in[4];
    const float* vw_e  = (const float*)d_in[5];
    const float* Wq = (const float*)d_in[6],  *bq = (const float*)d_in[7];
    const float* Wk = (const float*)d_in[8],  *bk = (const float*)d_in[9];
    const float* Wv = (const float*)d_in[10], *bv = (const float*)d_in[11];
    const float* Wo = (const float*)d_in[12], *bo = (const float*)d_in[13];
    const float* ln1g = (const float*)d_in[14], *ln1b = (const float*)d_in[15];
    const float* ln2g = (const float*)d_in[16], *ln2b = (const float*)d_in[17];
    const float* fw1 = (const float*)d_in[18], *fb1 = (const float*)d_in[19];
    const float* fw2 = (const float*)d_in[20], *fb2 = (const float*)d_in[21];
    const float* pw1 = (const float*)d_in[22], *pb1 = (const float*)d_in[23];
    const float* pw2 = (const float*)d_in[24], *pb2 = (const float*)d_in[25];
    const float* pw3 = (const float*)d_in[26], *pb3 = (const float*)d_in[27];
    // d_in[28..33] = de_w1/de_b1/de_w2/de_b2/mp_w/mp_b : dead w.r.t. outputs
    const float* dw1 = (const float*)d_in[34], *db1 = (const float*)d_in[35];
    const float* dw2 = (const float*)d_in[36], *db2 = (const float*)d_in[37];
    const float* dw3 = (const float*)d_in[38], *db3 = (const float*)d_in[39];
    const float* cw  = (const float*)d_in[40], *cb  = (const float*)d_in[41];

    float* out = (float*)d_out;
    float* out_coords = out;                     // B*C*N*2
    float* out_conf   = out + (size_t)ROWS * 2;  // B*C*N
    float* out_depth  = out_conf + ROWS;         // B*C*N

    // workspace layout (~344 MB)
    float* bufQ = (float*)d_ws;                       // ROWS*D   (query activations)
    float* bufA = bufQ + (size_t)ROWS * DD;           // ROWS*D   (qh / attn_out / P1+Q1)
    float* bufB = bufA + (size_t)ROWS * DD;           // ROWS*D   (context / P2+Q2)
    float* bufK = bufB + (size_t)ROWS * DD;           // MROWS*D
    float* bufV = bufK + (size_t)MROWS * DD;          // MROWS*D
    float* bufH = bufV + (size_t)MROWS * DD;          // FFN_CHUNK*DFF

    // 1) build queries
    k_build_q<<<(ROWS * DD) / 256, 256, 0, stream>>>(vw_e, kp_e, bufQ);

    // 2) transformer layers
    for (int l = 0; l < LL; ++l) {
        const float* wq = Wq + (size_t)l * DD * DD;
        const float* wk = Wk + (size_t)l * DD * DD;
        const float* wv = Wv + (size_t)l * DD * DD;
        const float* wo = Wo + (size_t)l * DD * DD;

        // qh = q @ Wq + bq
        k_gemm<<<dim3(DD / 64, ROWS / 128), 256, 0, stream>>>(
            bufQ, wq, bq + l * DD, bufA, ROWS, DD, DD, 0);
        // kh, vh (memory side, M = 2048)
        k_gemm<<<dim3(DD / 64, MROWS / 128), 256, 0, stream>>>(
            enc, wk, bk + l * DD, bufK, MROWS, DD, DD, 0);
        k_gemm<<<dim3(DD / 64, MROWS / 128), 256, 0, stream>>>(
            enc, wv, bv + l * DD, bufV, MROWS, DD, DD, 0);

        // softmax attention over 4 keys -> context in bufB
        k_attn<<<(BB * CN * HH) / 8, 256, 0, stream>>>(bufA, bufK, bufV, bufB);

        // attn_out = context @ Wo + bo
        k_gemm<<<dim3(DD / 64, ROWS / 128), 256, 0, stream>>>(
            bufB, wo, bo + l * DD, bufA, ROWS, DD, DD, 0);

        // q = LN(q + attn_out)
        k_res_ln<<<ROWS / 8, 256, 0, stream>>>(bufQ, bufA, ln1g + l * DD, ln1b + l * DD, ROWS);

        // FFN, chunked over M so hidden scratch stays bounded
        const float* w1 = fw1 + (size_t)l * DD * DFF;
        const float* w2 = fw2 + (size_t)l * DFF * DD;
        for (int m0 = 0; m0 < ROWS; m0 += FFN_CHUNK) {
            k_gemm<<<dim3(DFF / 64, FFN_CHUNK / 128), 256, 0, stream>>>(
                bufQ + (size_t)m0 * DD, w1, fb1 + l * DFF, bufH,
                FFN_CHUNK, DFF, DD, 1);  // GELU
            k_gemm<<<dim3(DD / 64, FFN_CHUNK / 128), 256, 0, stream>>>(
                bufH, w2, fb2 + l * DD, bufA + (size_t)m0 * DD,
                FFN_CHUNK, DD, DFF, 0);
        }

        // q = LN(q + ffn_out)
        k_res_ln<<<ROWS / 8, 256, 0, stream>>>(bufQ, bufA, ln2g + l * DD, ln2b + l * DD, ROWS);
    }

    // 3) heads
    float* P1 = bufA;                       // ROWS*D2
    float* Q1 = bufA + (size_t)ROWS * D2;   // ROWS*D2
    float* P2 = bufB;                       // ROWS*D4
    float* Q2 = bufB + (size_t)ROWS * D4;   // ROWS*D4

    k_gemm<<<dim3(D2 / 64, ROWS / 128), 256, 0, stream>>>(bufQ, pw1, pb1, P1, ROWS, D2, DD, 1);
    k_gemm<<<dim3(D4 / 64, ROWS / 128), 256, 0, stream>>>(P1, pw2, pb2, P2, ROWS, D4, D2, 1);
    k_gemm<<<dim3(D2 / 64, ROWS / 128), 256, 0, stream>>>(bufQ, dw1, db1, Q1, ROWS, D2, DD, 1);
    k_gemm<<<dim3(D4 / 64, ROWS / 128), 256, 0, stream>>>(Q1, dw2, db2, Q2, ROWS, D4, D2, 1);

    k_final<<<ROWS / 8, 256, 0, stream>>>(bufQ, P2, Q2, pw3, pb3, dw3, db3, cw, cb,
                                          okp, occ, out_coords, out_conf, out_depth);
}
